// FrameUpdate_87935160418330
// MI455X (gfx1250) — matrix operand order repaired and checked by reading, not compiled, not run on hardware
//
#include <hip/hip_runtime.h>
#include <hip/hip_bf16.h>

typedef __attribute__((ext_vector_type(16))) _Float16 v16h;
typedef __attribute__((ext_vector_type(8)))  _Float16 v8h;
typedef __attribute__((ext_vector_type(8)))  float    v8f;
typedef __attribute__((ext_vector_type(4)))  unsigned v4u;
typedef __attribute__((ext_vector_type(8)))  unsigned v8u;

namespace {
constexpr int NN   = 8192;
constexpr int KNNK = 60;
constexpr int EKNN = NN * KNNK;     // 491520
constexpr int ESEQ = NN * 4;        // 32768
constexpr int ETOT = EKNN + ESEQ;   // 524288
}

__device__ __forceinline__ int edge_id(int node, int el) {
  return (el < KNNK) ? node * KNNK + el : EKNN + node * 4 + (el - KNNK);
}

// ---------------------------------------------------------------------------
// TDM: async 2-D tile load (tile == tensor) global -> LDS via descriptor in
// SGPRs. D# group0 (128b): count=1 | lds_addr | global_addr[56:0] | type=2.
// D# group1 (256b): data_size=4B, tensor_dim0/1, tile_dim0/1, dim0 stride.
// ---------------------------------------------------------------------------
__device__ __forceinline__ void tdm_load_2d_f32(const void* gptr, void* ldsptr,
                                                unsigned dim0, unsigned dim1) {
  unsigned long long ga = (unsigned long long)gptr;
  unsigned la = (unsigned)(unsigned long long)ldsptr;   // LDS byte offset
  v4u g0;
  g0[0] = 1u;                                              // count=1 (valid, user)
  g0[1] = la;                                              // lds_addr [63:32]
  g0[2] = (unsigned)ga;                                    // global_addr [95:64]
  g0[3] = ((unsigned)(ga >> 32) & 0x01FFFFFFu) | (2u << 30); // addr[56:32] | type=2
  v8u g1;
  g1[0] = (2u << 16);                                      // wg_mask=0, data_size=4B
  g1[1] = (dim0 & 0xFFFFu) << 16;                          // tensor_dim0 lo16
  g1[2] = ((dim0 >> 16) & 0xFFFFu) | ((dim1 & 0xFFFFu) << 16);   // dim0 hi | dim1 lo
  g1[3] = ((dim1 >> 16) & 0xFFFFu) | ((dim0 & 0xFFFFu) << 16);   // dim1 hi | tile_dim0
  g1[4] = dim1 & 0xFFFFu;                                  // tile_dim1, tile_dim2=0
  g1[5] = dim0;                                            // tensor_dim0_stride lo32
  g1[6] = 0u;
  g1[7] = 0u;
  asm volatile("tensor_load_to_lds %0, %1" :: "s"(g0), "s"(g1) : "memory");
}

// ---------------------------------------------------------------------------
// 1. KNN: one block per row; D2 row in LDS (32 KB), 60 iterative argmins.
// ---------------------------------------------------------------------------
__global__ __launch_bounds__(256) void knn_kernel(
    const float* __restrict__ X, const int* __restrict__ batch,
    const unsigned char* __restrict__ xmask, int* __restrict__ nbr) {
  __shared__ float sD[NN];
  __shared__ float sV[256];
  __shared__ int   sI[256];
  const int i = blockIdx.x, tid = threadIdx.x;
  const float xi0 = X[i*3+0], xi1 = X[i*3+1], xi2 = X[i*3+2];
  const int bi = batch[i];
  const float INF = __builtin_inff();
  for (int j = tid; j < NN; j += 256) {
    float dx = X[j*3+0]-xi0, dy = X[j*3+1]-xi1, dz = X[j*3+2]-xi2;
    float d2 = dx*dx + dy*dy + dz*dz;
    bool bad = (j == i) || (batch[j] != bi) || (xmask[j] == 0);
    sD[j] = bad ? INF : d2;
  }
  __syncthreads();
  for (int k = 0; k < KNNK; ++k) {
    float bv = INF; int bidx = 0x7fffffff;
    for (int j = tid; j < NN; j += 256) {
      float v = sD[j];
      if (v < bv || (v == bv && j < bidx)) { bv = v; bidx = j; }
    }
    sV[tid] = bv; sI[tid] = bidx;
    __syncthreads();
    for (int s = 128; s > 0; s >>= 1) {
      if (tid < s) {
        float ov = sV[tid+s]; int oi = sI[tid+s];
        if (ov < sV[tid] || (ov == sV[tid] && oi < sI[tid])) { sV[tid]=ov; sI[tid]=oi; }
      }
      __syncthreads();
    }
    if (tid == 0) { int sel = sI[0]; nbr[i*KNNK + k] = sel; sD[sel] = INF; }
    __syncthreads();
  }
}

// ---------------------------------------------------------------------------
// 2. Wa1 (102x128 f32) -> f16 pre-swizzled into WMMA B-fragment order:
//    [nt(8)][kc(4)][lane(32)][16 halfs contiguous] so each lane loads 32 B.
//    B lane layout: col = lane&15; K = kc*32 + (lane>>4)*16 + t.
// ---------------------------------------------------------------------------
__global__ void cvt_wa1_kernel(const float* __restrict__ Wa1, _Float16* __restrict__ w1p) {
  int t = blockIdx.x * 256 + threadIdx.x;    // 16384
  int frag = t >> 9;          // nt*4 + kc
  int rem  = t & 511;
  int lane = rem >> 4;
  int h    = rem & 15;
  int nt = frag >> 2, kc = frag & 3;
  int k = kc * 32 + (lane >> 4) * 16 + h;
  int n = nt * 16 + (lane & 15);
  float v = (k < 102) ? Wa1[k * 128 + n] : 0.f;
  w1p[t] = (_Float16)v;
}

// ---------------------------------------------------------------------------
// 3. Per-edge geometry + features: z row (f16, padded 128), valid flag, D1.
// ---------------------------------------------------------------------------
__global__ __launch_bounds__(256) void edge_kernel(
    const float* __restrict__ X, const int* __restrict__ batch,
    const unsigned char* __restrict__ xmask, const unsigned char* __restrict__ nmask,
    const float* __restrict__ nf, const int* __restrict__ seq,
    const int* __restrict__ nbr, _Float16* __restrict__ zbuf,
    float* __restrict__ validb, float* __restrict__ d1b) {
  int e = blockIdx.x * 256 + threadIdx.x;
  if (e >= ETOT) return;
  int src, dst;
  if (e < EKNN) { dst = e / KNNK; src = nbr[e]; }
  else { int s = e - EKNN; src = seq[s]; dst = seq[ESEQ + s]; }

  float vx = X[src*3+0]-X[dst*3+0], vy = X[src*3+1]-X[dst*3+1], vz = X[src*3+2]-X[dst*3+2];
  float dist = sqrtf(vx*vx + vy*vy + vz*vz + 1e-12f);
  bool valid = (dist > 0.001f) && (batch[src] == batch[dst]) && xmask[src] && xmask[dst];
  validb[e] = valid ? 1.f : 0.f;
  float inv = 1.f / dist;
  float hx = vx*inv, hy = vy*inv, hz = vz*inv;
  float rx, ry, rz;
  if (fabsf(hz) < 0.99f) { rx = 0.f; ry = 0.f; rz = 1.f; } else { rx = 1.f; ry = 0.f; rz = 0.f; }
  float zx = hy*rz - hz*ry, zy = hz*rx - hx*rz, zz = hx*ry - hy*rx;
  float zn = 1.f / (sqrtf(zx*zx + zy*zy + zz*zz) + 1e-9f);
  zx *= zn; zy *= zn; zz *= zn;
  float yx = zy*hz - zz*hy, yy = zz*hx - zx*hz, yz = zx*hy - zy*hx;
  float R[3][3] = {{zx,zy,zz},{hx,hy,hz},{yx,yy,yz}};
  const int p[3] = {1,2,0};
  #pragma unroll
  for (int a = 0; a < 3; ++a)
    #pragma unroll
    for (int l = 0; l < 3; ++l)
      d1b[(size_t)e*9 + a*3 + l] = R[p[a]][p[l]];

  _Float16* zp = zbuf + (size_t)e * 128;
  float edt_s = (nmask[src] && !xmask[src]) ? 1.f : 0.f;
  float edt_d = (nmask[dst] && !xmask[dst]) ? 1.f : 0.f;
  const float4* nfs = (const float4*)(nf + (size_t)src * 128);
  const float4* nfd = (const float4*)(nf + (size_t)dst * 128);
  #pragma unroll
  for (int t = 0; t < 8; ++t) {
    float4 a = nfs[t];
    zp[t*4+0] = (_Float16)a.x; zp[t*4+1] = (_Float16)a.y;
    zp[t*4+2] = (_Float16)a.z; zp[t*4+3] = (_Float16)a.w;
  }
  zp[32] = (_Float16)0.f; zp[33] = (_Float16)0.f; zp[34] = (_Float16)edt_s;
  #pragma unroll
  for (int t = 0; t < 8; ++t) {
    float4 a = nfd[t];
    zp[35+t*4+0] = (_Float16)a.x; zp[35+t*4+1] = (_Float16)a.y;
    zp[35+t*4+2] = (_Float16)a.z; zp[35+t*4+3] = (_Float16)a.w;
  }
  zp[67] = (_Float16)0.f; zp[68] = (_Float16)0.f; zp[69] = (_Float16)edt_d;
  #pragma unroll
  for (int k = 0; k < 16; ++k) {
    float t = (dist - (20.f/15.f)*k) * 0.8f;        // sigma = 1.25
    zp[70 + k] = (_Float16)expf(-t*t);
  }
  float off = (float)(src - dst);
  #pragma unroll
  for (int k = 0; k < 8; ++k) {
    float f = expf(-(float)k * 0.125f * 9.210340371976184f); // 10000^(-k/8)
    float a = off * f;
    zp[86 + k] = (_Float16)cosf(a);
    zp[94 + k] = (_Float16)sinf(a);
  }
  #pragma unroll
  for (int t = 102; t < 128; ++t) zp[t] = (_Float16)0.f;
}

// ---------------------------------------------------------------------------
// 4. WMMA: hid = relu(z @ Wa1 + ba1); logits[h] = hid_tile . wa2[h]; mask.
//    4 waves/block, each wave owns one 16-edge tile; 32 v_wmma per wave.
//    A loaded as 2x b128 per fragment; B pre-swizzled -> contiguous 32 B.
// ---------------------------------------------------------------------------
__global__ __launch_bounds__(128) void attn_logits_kernel(
    const _Float16* __restrict__ zbuf, const _Float16* __restrict__ w1p,
    const float* __restrict__ ba1, const float* __restrict__ wa2,
    const float* __restrict__ validb, float* __restrict__ logit) {
  const int wave = threadIdx.x >> 5;
  const int lane = threadIdx.x & 31;
  const int tile = blockIdx.x * 4 + wave;
  const int m0 = tile * 16;
  const int halfsel = lane >> 4;
  const int ln = lane & 15;

  const _Float16* arow = zbuf + (size_t)(m0 + ln) * 128;
  __builtin_prefetch(arow + 64 * 128, 0, 0);   // stream-ahead -> global_prefetch

  v16h afr[4];
  #pragma unroll
  for (int kc = 0; kc < 4; ++kc) {
    const int kb = kc * 32 + halfsel * 16;
    v8h alo = *(const v8h*)(arow + kb);
    v8h ahi = *(const v8h*)(arow + kb + 8);
    afr[kc] = __builtin_shufflevector(alo, ahi, 0,1,2,3,4,5,6,7,8,9,10,11,12,13,14,15);
  }

  #pragma unroll
  for (int nt = 0; nt < 8; ++nt) {
    v8f acc = {0.f,0.f,0.f,0.f,0.f,0.f,0.f,0.f};
    #pragma unroll
    for (int kc = 0; kc < 4; ++kc) {
      v16h b = *(const v16h*)(w1p + ((((nt * 4 + kc) * 32) + lane) << 4));
      acc = __builtin_amdgcn_wmma_f32_16x16x32_f16(false, afr[kc], false, b,
                                                   (short)0, acc, false, false);
    }
    const float bias = ba1[nt * 16 + ln];
    const float w2   = wa2[nt * 16 + ln];
    float pr[8];
    #pragma unroll
    for (int r = 0; r < 8; ++r) { float h = acc[r] + bias; pr[r] = fmaxf(h, 0.f) * w2; }
    #pragma unroll
    for (int r = 0; r < 8; ++r)
      for (int off = 8; off > 0; off >>= 1) pr[r] += __shfl_xor(pr[r], off, 32);
    if (ln == 0) {
      #pragma unroll
      for (int r = 0; r < 8; ++r) {
        int e = m0 + r + 8 * halfsel;
        logit[(size_t)e * 8 + nt] = (validb[e] > 0.f) ? pr[r] : -1e9f;
      }
    }
  }
}

// ---------------------------------------------------------------------------
// 5. Segment softmax: one wave per node over its 64 edges (60 kNN + 4 seq).
// ---------------------------------------------------------------------------
__global__ __launch_bounds__(256) void softmax_kernel(
    float* __restrict__ logit, const float* __restrict__ validb) {
  const int node = blockIdx.x * 8 + (threadIdx.x >> 5);
  const int lane = threadIdx.x & 31;
  const int e1 = edge_id(node, lane);
  const int e2 = edge_id(node, lane + 32);
  const float v1 = validb[e1], v2 = validb[e2];
  #pragma unroll
  for (int h = 0; h < 8; ++h) {
    float l1 = logit[(size_t)e1*8 + h], l2 = logit[(size_t)e2*8 + h];
    float m = fmaxf(l1, l2);
    for (int off = 16; off > 0; off >>= 1) m = fmaxf(m, __shfl_xor(m, off, 32));
    float x1 = (v1 > 0.f) ? expf(l1 - m) : 0.f;
    float x2 = (v2 > 0.f) ? expf(l2 - m) : 0.f;
    float s = x1 + x2;
    for (int off = 16; off > 0; off >>= 1) s += __shfl_xor(s, off, 32);
    float inv = 1.f / (s + 1e-9f);
    logit[(size_t)e1*8 + h] = x1 * inv;
    logit[(size_t)e2*8 + h] = x2 * inv;
  }
}

// ---------------------------------------------------------------------------
// 6. Message aggregation + Wo: one block per node. Wv/Wo staged into LDS by
//    the Tensor Data Mover (tensor_load_to_lds + s_wait_tensorcnt).
// ---------------------------------------------------------------------------
__global__ __launch_bounds__(256) void message_kernel(
    const int* __restrict__ nbr, const int* __restrict__ seq,
    const float* __restrict__ alpha, const float* __restrict__ d1b,
    const float* __restrict__ nf, const float* __restrict__ rots,
    const unsigned char* __restrict__ xmask, const unsigned char* __restrict__ nmask,
    const float* __restrict__ Wv, const float* __restrict__ Wo,
    float* __restrict__ ybuf) {
  __shared__ float sWv[35*64];
  __shared__ float sWo[64*32];
  __shared__ float sXr[8][4*36];
  __shared__ float sAl[64*8];
  __shared__ float sD1[64*9];
  __shared__ int   sSrc[64];
  __shared__ float sPart[8][256];
  __shared__ float sAgg[256];

  const int i = blockIdx.x, tid = threadIdx.x;
  const int w = tid >> 5, lane = tid & 31;

  if (tid < 32) {                       // one wave drives the TDM (EXEC ignored)
    tdm_load_2d_f32(Wv, &sWv[0], 64u, 35u);   // 35 rows x 64 ch, stride 64
    tdm_load_2d_f32(Wo, &sWo[0], 32u, 64u);   // 64 rows x 32 ch, stride 32
  }
  if (tid < 64) {
    const int el = tid;
    const int e  = edge_id(i, el);
    sSrc[el] = (el < KNNK) ? nbr[i*KNNK + el] : seq[i*4 + (el - KNNK)];
    #pragma unroll
    for (int h = 0; h < 8; ++h) sAl[el*8 + h] = alpha[(size_t)e*8 + h];
    #pragma unroll
    for (int k = 0; k < 9; ++k) sD1[el*9 + k] = d1b[(size_t)e*9 + k];
  }
  __builtin_amdgcn_s_wait_tensorcnt(0);
  __syncthreads();

  const float IDE[3][3] = {{-0.525f,1.363f,0.f},{0.f,0.f,0.f},{1.526f,0.f,0.f}};
  float acc[4][2] = {{0.f,0.f},{0.f,0.f},{0.f,0.f},{0.f,0.f}};

  for (int step = 0; step < 8; ++step) {
    const int el = step * 8 + w;
    const int s  = sSrc[el];
    const float edt = (nmask[s] && !xmask[s]) ? 1.f : 0.f;
    for (int idx = lane; idx < 140; idx += 32) {
      int m = idx / 35, c = idx % 35;
      float v;
      if (c < 32)      v = nf[(size_t)s*128 + m*32 + c];
      else if (m == 0) v = (c == 34) ? edt : 0.f;
      else {
        int a = c - 32, r = m - 1;
        v = rots[(size_t)s*9 + r*3 + 0]*IDE[a][0]
          + rots[(size_t)s*9 + r*3 + 1]*IDE[a][1]
          + rots[(size_t)s*9 + r*3 + 2]*IDE[a][2];
      }
      sXr[w][m*36 + c] = v;
    }
    __syncthreads();
    const float* D = &sD1[el*9];
    #pragma unroll
    for (int cc = 0; cc < 2; ++cc) {
      const int ch = lane + cc * 32;
      float vr0 = 0.f, vr1 = 0.f, vr2 = 0.f, vr3 = 0.f;
      for (int c = 0; c < 35; ++c) {
        float wv = sWv[c*64 + ch];
        vr0 += sXr[w][0*36 + c] * wv;
        vr1 += sXr[w][1*36 + c] * wv;
        vr2 += sXr[w][2*36 + c] * wv;
        vr3 += sXr[w][3*36 + c] * wv;
      }
      float t1 = D[0]*vr1 + D[1]*vr2 + D[2]*vr3;   // x_src rows = D1 @ raw
      float t2 = D[3]*vr1 + D[4]*vr2 + D[5]*vr3;
      float t3 = D[6]*vr1 + D[7]*vr2 + D[8]*vr3;
      const float a = sAl[el*8 + (ch >> 3)];
      float a0 = a*vr0, a1 = a*t1, a2 = a*t2, a3 = a*t3;
      acc[0][cc] += a0;                              // msg rows = D1^T @ (alpha*v)
      acc[1][cc] += D[0]*a1 + D[3]*a2 + D[6]*a3;
      acc[2][cc] += D[1]*a1 + D[4]*a2 + D[7]*a3;
      acc[3][cc] += D[2]*a1 + D[5]*a2 + D[8]*a3;
    }
    __syncthreads();
  }

  #pragma unroll
  for (int m = 0; m < 4; ++m) {
    sPart[w][m*64 + lane]      = acc[m][0];
    sPart[w][m*64 + lane + 32] = acc[m][1];
  }
  __syncthreads();
  float tot = 0.f;
  #pragma unroll
  for (int ww = 0; ww < 8; ++ww) tot += sPart[ww][tid];
  sAgg[tid] = tot;
  __syncthreads();
  if (tid < 128) {
    int m = tid >> 5, c = tid & 31;
    float y = nf[(size_t)i*128 + m*32 + c];
    for (int k = 0; k < 64; ++k) y += sAgg[m*64 + k] * sWo[k*32 + c];
    ybuf[(size_t)i*128 + m*32 + c] = y;
  }
}

// ---------------------------------------------------------------------------
// 7. Node update MLPs + quaternion frame update: one wave per node, lane=chan.
// ---------------------------------------------------------------------------
__global__ __launch_bounds__(256) void node_update_kernel(
    const float* __restrict__ ybuf, const float* __restrict__ rots,
    const float* __restrict__ trans, const unsigned char* __restrict__ nmask,
    const float* __restrict__ Wf1, const float* __restrict__ bf1,
    const float* __restrict__ Wf2, const float* __restrict__ Wg,
    const float* __restrict__ Wr1, const float* __restrict__ br1,
    const float* __restrict__ Wr2, const float* __restrict__ br2,
    const float* __restrict__ Wr3, const float* __restrict__ br3,
    const float* __restrict__ wt, float* __restrict__ out) {
  const int i = blockIdx.x * 8 + (threadIdx.x >> 5);
  const int c = threadIdx.x & 31;
  const float y0 = ybuf[(size_t)i*128 + c];
  const float y1 = ybuf[(size_t)i*128 + 32 + c];
  const float y2 = ybuf[(size_t)i*128 + 64 + c];
  const float y3 = ybuf[(size_t)i*128 + 96 + c];
  const float s0 = y0;
  float h = bf1[c];
  for (int j = 0; j < 32; ++j) h += __shfl(s0, j, 32) * Wf1[j*32 + c];
  h = fmaxf(h, 0.f);
  float s0n = s0, g = 0.f;
  for (int j = 0; j < 32; ++j) { float t = __shfl(h, j, 32); s0n += t*Wf2[j*32 + c]; g += t*Wg[j*32 + c]; }
  g = 1.f / (1.f + expf(-g));
  const float bb0 = s0n, bb1 = y1*g, bb2 = y2*g, bb3 = y3*g;
  float ra = br1[c], rb = br1[32 + c];
  for (int j = 0; j < 32; ++j) { float t = __shfl(bb0, j, 32); ra += t*Wr1[j*64 + c]; rb += t*Wr1[j*64 + 32 + c]; }
  ra = fmaxf(ra, 0.f); rb = fmaxf(rb, 0.f);
  float r2 = br2[c];
  for (int j = 0; j < 32; ++j)
    r2 += __shfl(ra, j, 32)*Wr2[j*32 + c] + __shfl(rb, j, 32)*Wr2[(32 + j)*32 + c];
  r2 = fmaxf(r2, 0.f);
  float q1 = br3[0], q2 = br3[1], q3 = br3[2];
  for (int j = 0; j < 32; ++j) { float t = __shfl(r2, j, 32); q1 += t*Wr3[j*6+0]; q2 += t*Wr3[j*6+1]; q3 += t*Wr3[j*6+2]; }
  const float qn = rsqrtf(1.f + q1*q1 + q2*q2 + q3*q3);
  const float qw = qn, qx = q1*qn, qy = q2*qn, qz = q3*qn;
  float U[3][3];
  U[0][0]=1.f-2.f*(qy*qy+qz*qz); U[0][1]=2.f*(qx*qy-qw*qz); U[0][2]=2.f*(qx*qz+qw*qy);
  U[1][0]=2.f*(qx*qy+qw*qz); U[1][1]=1.f-2.f*(qx*qx+qz*qz); U[1][2]=2.f*(qy*qz-qw*qx);
  U[2][0]=2.f*(qx*qz-qw*qy); U[2][1]=2.f*(qy*qz+qw*qx); U[2][2]=1.f-2.f*(qx*qx+qy*qy);
  float Rm[9];
  #pragma unroll
  for (int k = 0; k < 9; ++k) Rm[k] = rots[(size_t)i*9 + k];
  float RU[9];
  #pragma unroll
  for (int r = 0; r < 3; ++r)
    #pragma unroll
    for (int cc2 = 0; cc2 < 3; ++cc2)
      RU[r*3+cc2] = Rm[r*3+0]*U[0][cc2] + Rm[r*3+1]*U[1][cc2] + Rm[r*3+2]*U[2][cc2];
  float t0 = bb1*wt[c], t1 = bb2*wt[c], t2 = bb3*wt[c];
  for (int off = 16; off > 0; off >>= 1) {
    t0 += __shfl_xor(t0, off, 32); t1 += __shfl_xor(t1, off, 32); t2 += __shfl_xor(t2, off, 32);
  }
  const float mn = nmask[i] ? 1.f : 0.f;
  float* op = out + (size_t)i * 140;
  if (c < 9)  op[c] = mn*RU[c] + (1.f - mn)*Rm[c];
  if (c >= 9 && c < 12) {
    int d = c - 9;
    float tu = (d == 0) ? t0 : ((d == 1) ? t1 : t2);
    op[c] = trans[i*3 + d] + mn * tu;
  }
  op[12 +  0 + c] = bb0;
  op[12 + 32 + c] = bb1;
  op[12 + 64 + c] = bb2;
  op[12 + 96 + c] = bb3;
}

// ---------------------------------------------------------------------------
extern "C" void kernel_launch(void* const* d_in, const int* in_sizes, int n_in,
                              void* d_out, int out_size, void* d_ws, size_t ws_size,
                              hipStream_t stream) {
  const float* rots  = (const float*)d_in[0];
  const float* trans = (const float*)d_in[1];
  const float* nf    = (const float*)d_in[2];
  const int*   batch = (const int*)d_in[3];
  const unsigned char* xmask = (const unsigned char*)d_in[4];
  const unsigned char* nmask = (const unsigned char*)d_in[5];
  const int*   seq   = (const int*)d_in[6];
  const float* Wa1 = (const float*)d_in[7];
  const float* ba1 = (const float*)d_in[8];
  const float* wa2 = (const float*)d_in[9];
  const float* Wv  = (const float*)d_in[10];
  const float* Wo  = (const float*)d_in[11];
  const float* Wf1 = (const float*)d_in[12];
  const float* bf1 = (const float*)d_in[13];
  const float* Wf2 = (const float*)d_in[14];
  const float* Wg  = (const float*)d_in[15];
  const float* Wr1 = (const float*)d_in[16];
  const float* br1 = (const float*)d_in[17];
  const float* Wr2 = (const float*)d_in[18];
  const float* br2 = (const float*)d_in[19];
  const float* Wr3 = (const float*)d_in[20];
  const float* br3 = (const float*)d_in[21];
  const float* wt  = (const float*)d_in[22];
  float* out = (float*)d_out;
  (void)in_sizes; (void)n_in; (void)out_size; (void)ws_size;

  char* ws = (char*)d_ws;
  size_t o = 0;
  auto take = [&](size_t bytes) -> char* {
    char* p = ws + o; o = (o + bytes + 255) & ~size_t(255); return p;
  };
  int*      nbr    = (int*)take(sizeof(int) * (size_t)NN * KNNK);
  _Float16* zbuf   = (_Float16*)take(sizeof(_Float16) * (size_t)ETOT * 128);
  float*    validb = (float*)take(sizeof(float) * (size_t)ETOT);
  float*    d1b    = (float*)take(sizeof(float) * (size_t)ETOT * 9);
  float*    albuf  = (float*)take(sizeof(float) * (size_t)ETOT * 8);
  _Float16* w1p    = (_Float16*)take(sizeof(_Float16) * 128 * 128);
  float*    ybuf   = (float*)take(sizeof(float) * (size_t)NN * 128);

  knn_kernel<<<NN, 256, 0, stream>>>(trans, batch, xmask, nbr);
  cvt_wa1_kernel<<<64, 256, 0, stream>>>(Wa1, w1p);
  edge_kernel<<<ETOT/256, 256, 0, stream>>>(trans, batch, xmask, nmask, nf, seq, nbr,
                                            zbuf, validb, d1b);
  attn_logits_kernel<<<ETOT/64, 128, 0, stream>>>(zbuf, w1p, ba1, wa2, validb, albuf);
  softmax_kernel<<<NN/8, 256, 0, stream>>>(albuf, validb);
  message_kernel<<<NN, 256, 0, stream>>>(nbr, seq, albuf, d1b, nf, rots, xmask, nmask,
                                         Wv, Wo, ybuf);
  node_update_kernel<<<NN/8, 256, 0, stream>>>(ybuf, rots, trans, nmask,
                                               Wf1, bf1, Wf2, Wg, Wr1, br1, Wr2, br2,
                                               Wr3, br3, wt, out);
}
